// gated_TPP_48593259987404
// MI455X (gfx1250) — matrix operand order repaired and checked by reading, not compiled, not run on hardware
//
#include <hip/hip_runtime.h>
#include <cstdint>

typedef _Float16 v16h __attribute__((ext_vector_type(16)));
typedef _Float16 v8h  __attribute__((ext_vector_type(8)));
typedef float    v8f  __attribute__((ext_vector_type(8)));
typedef float    v4f  __attribute__((ext_vector_type(4)));

#define B_    16
#define L_    2048
#define D_    256
#define LP_   2080   // padded embT row stride (multiple of 16, covers j-tile overhang)
#define TMAXI (1.0f/200.0f)
#define LNEPS 1e-6f

// ---------------- kernel 1: transposed f16 time-encoding ----------------
// embT[b][d][j] = (d even ? sin : cos)(t_j * 10000^(-2d/D)); j >= L padded w/ 0
__global__ void embT_kernel(const float* __restrict__ et,
                            _Float16* __restrict__ embT) {
  int bd = blockIdx.x;                 // 0 .. B*D-1
  int b  = bd >> 8;
  int d  = bd & 255;
  float invpv = __expf(-2.0f * (float)d * (1.0f / (float)D_) * 9.210340371976184f); // ln(1e4)
  _Float16* row = embT + (size_t)bd * LP_;
  const float* trow = et + (size_t)b * L_;
  for (int j = threadIdx.x; j < LP_; j += blockDim.x) {
    float v = 0.0f;
    if (j < L_) {
      float x = trow[j] * invpv;
      v = (d & 1) ? cosf(x) : sinf(x);
    }
    row[j] = (_Float16)v;
  }
}

// ---------------- kernel 2: weights f32 -> f16 ----------------
__global__ void wcvt_kernel(const float* __restrict__ Wn, const float* __restrict__ Wi,
                            _Float16* __restrict__ WnH, _Float16* __restrict__ WiH) {
  int idx = blockIdx.x * blockDim.x + threadIdx.x;
  if (idx < D_ * D_) {
    WnH[idx] = (_Float16)Wn[idx];
    WiH[idx] = (_Float16)Wi[idx];
  }
}

// ---------------- kernel 3: fused scores@emb + LN + generator ----------------
__launch_bounds__(256, 2)
__global__ void fused_kernel(const float* __restrict__ et,
                             const float* __restrict__ noise,
                             const float* __restrict__ gp,
                             const float* __restrict__ lsc,
                             const _Float16* __restrict__ embT,
                             const _Float16* __restrict__ WnH,
                             const _Float16* __restrict__ WiH,
                             const float* __restrict__ w_time,
                             const float* __restrict__ b_time,
                             const float* __restrict__ gamma,
                             const float* __restrict__ beta,
                             float* __restrict__ out) {
  __shared__ __align__(16) _Float16 sA[2][16 * 40];  // double-buffered score tile
  __shared__ __align__(16) float    sHid[16 * 260];  // hidden tile f32
  __shared__ __align__(16) _Float16 sH16[16 * 264];  // normalized hidden f16
  __shared__ float sGamma[256], sBeta[256], sTi[16];
  __shared__ float sPart[256], sPart2[256], sMu[16], sRs[16];
  __shared__ float sOutW[8 * 16];

  const int tid   = threadIdx.x;
  const int wv    = tid >> 5;        // wave 0..7
  const int ln    = tid & 31;        // lane
  const int lnlow = ln & 15;
  const int hi    = ln >> 4;         // lane half
  const int kb8   = hi << 3;         // unused; A-frag K base is kb16 too
  const int kb16  = hi << 4;         // B-frag K base
  const int row16 = lnlow;           // A-frag row

  const int b  = blockIdx.x >> 7;    // batch
  const int it = blockIdx.x & 127;   // i-tile
  const int i0 = it << 4;

  // uniform scalar params
  const float gl      = 1.0f / (1.0f + __expf(-gp[0]));
  const float gs      = 1.0f / (1.0f + __expf(-gp[1]));
  const float inv_s2  = 2.0f / gs;   // folded *2 for the tanh identity
  const float lsv     = lsc[0];
  const float sp_ls   = (lsv > 20.0f) ? lsv : log1pf(__expf(lsv));
  const float inv_ls2 = 1.0f / (sp_ls * sp_ls);
  const float* __restrict__ et_row = et + (size_t)b * L_;

  sGamma[tid] = gamma[tid];
  sBeta[tid]  = beta[tid];
  if (tid < 16) sTi[tid] = et_row[i0 + tid];
  __syncthreads();

  // branchless score tile: 1+tanh(y) = 2 - 2/(exp(2y)+1), causal mask via select
  auto compute_scores = [&](int kk, _Float16* __restrict__ buf) {
    const int j0 = kk << 5;
    #pragma unroll
    for (int e = tid; e < 512; e += 256) {
      int r = e >> 5, c = e & 31;
      int i = i0 + r, j = j0 + c;
      int jc = (j < L_) ? j : (L_ - 1);
      float dt   = fabsf(sTi[r] - et_row[jc]);
      float kv   = __expf(-dt * dt * inv_ls2);
      float e2   = __expf((dt * TMAXI - gl) * inv_s2);
      float gate = 2.0f - 2.0f * __builtin_amdgcn_rcpf(e2 + 1.0f);
      float v    = (j <= i) ? kv * gate : 0.0f;
      buf[r * 40 + c] = (_Float16)v;
    }
  };

  // ---- phase 1: hidden = scores @ emb (causal: only j-tiles <= diagonal) ----
  v8f acc0 = {0.f,0.f,0.f,0.f,0.f,0.f,0.f,0.f};
  v8f acc1 = {0.f,0.f,0.f,0.f,0.f,0.f,0.f,0.f};
  const int nk = ((i0 + 15) >> 5) + 1;
  const int col0 = (wv << 5) + lnlow;
  const int col1 = col0 + 16;
  const _Float16* __restrict__ embB = embT + (size_t)b * D_ * LP_;

  compute_scores(0, sA[0]);
  __syncthreads();

  for (int kk = 0; kk < nk; ++kk) {
    const int j0 = kk << 5;
    const _Float16* cur = sA[kk & 1];

    // A fragment from LDS (contiguous 16 K at kb16)
    const _Float16* ap = cur + row16 * 40 + kb16;
    v8h a0 = *(const v8h*)ap;
    v8h a1 = *(const v8h*)(ap + 8);
    // B fragments: contiguous 16 f16 from transposed emb
    const _Float16* bp0 = embB + (size_t)col0 * LP_ + j0 + kb16;
    const _Float16* bp1 = embB + (size_t)col1 * LP_ + j0 + kb16;
    v8h b00 = *(const v8h*)bp0, b01 = *(const v8h*)(bp0 + 8);
    v8h b10 = *(const v8h*)bp1, b11 = *(const v8h*)(bp1 + 8);

    // overlap next score tile with in-flight loads / WMMA issue
    if (kk + 1 < nk) compute_scores(kk + 1, sA[(kk + 1) & 1]);

    v16h A  = __builtin_shufflevector(a0,  a1,  0,1,2,3,4,5,6,7,8,9,10,11,12,13,14,15);
    v16h B0 = __builtin_shufflevector(b00, b01, 0,1,2,3,4,5,6,7,8,9,10,11,12,13,14,15);
    v16h B1 = __builtin_shufflevector(b10, b11, 0,1,2,3,4,5,6,7,8,9,10,11,12,13,14,15);
    acc0 = __builtin_amdgcn_wmma_f32_16x16x32_f16(false, A, false, B0, (short)0, acc0, false, false);
    acc1 = __builtin_amdgcn_wmma_f32_16x16x32_f16(false, A, false, B1, (short)0, acc1, false, false);
    __syncthreads();
  }

  // ---- phase 2: LayerNorm in LDS ----
  #pragma unroll
  for (int r = 0; r < 8; ++r) {
    int rr = r + (hi << 3);
    sHid[rr * 260 + col0] = acc0[r];
    sHid[rr * 260 + col1] = acc1[r];
  }
  __syncthreads();
  {
    int row = tid >> 4, seg = tid & 15;
    const float* hr = &sHid[row * 260 + (seg << 4)];
    float s1 = 0.f, s2 = 0.f;
    #pragma unroll
    for (int e = 0; e < 16; ++e) { float x = hr[e]; s1 += x; s2 += x * x; }
    sPart[tid] = s1; sPart2[tid] = s2;
  }
  __syncthreads();
  if (tid < 16) {
    float s1 = 0.f, s2 = 0.f;
    #pragma unroll
    for (int e = 0; e < 16; ++e) { s1 += sPart[tid * 16 + e]; s2 += sPart2[tid * 16 + e]; }
    float mu  = s1 * (1.0f / 256.0f);
    float var = s2 * (1.0f / 256.0f) - mu * mu;
    sMu[tid] = mu;
    sRs[tid] = rsqrtf(var + LNEPS);
  }
  __syncthreads();
  {
    int row = tid >> 4, seg = tid & 15;
    float mu = sMu[row], rs = sRs[row];
    #pragma unroll
    for (int e = 0; e < 16; ++e) {
      int c = (seg << 4) + e;
      sH16[row * 264 + c] =
          (_Float16)((sHid[row * 260 + c] - mu) * rs * sGamma[c] + sBeta[c]);
    }
  }
  __syncthreads();

  // ---- phase 3: h = relu(noise@Wn^T + hidden@Wi^T), out = softplus(h@w_time^T + b) ----
  v8f h0 = {0.f,0.f,0.f,0.f,0.f,0.f,0.f,0.f};
  v8f h1 = {0.f,0.f,0.f,0.f,0.f,0.f,0.f,0.f};
  const float* nbase = noise + ((size_t)(b * L_) + i0 + row16) * D_;

  #pragma unroll
  for (int kk = 0; kk < 8; ++kk) {
    const int k0 = kk << 5;
    // A from noise (f32 -> f16)
    const float* np = nbase + k0 + kb16;
    v4f f0 = *(const v4f*)np;
    v4f f1 = *(const v4f*)(np + 4);
    v4f f2 = *(const v4f*)(np + 8);
    v4f f3 = *(const v4f*)(np + 12);
    v16h An;
    #pragma unroll
    for (int e = 0; e < 4; ++e) {
      An[e]      = (_Float16)f0[e];
      An[4 + e]  = (_Float16)f1[e];
      An[8 + e]  = (_Float16)f2[e];
      An[12 + e] = (_Float16)f3[e];
    }
    // A from normalized hidden (LDS)
    const _Float16* hp = &sH16[row16 * 264 + k0 + kb16];
    v8h ha0 = *(const v8h*)hp;
    v8h ha1 = *(const v8h*)(hp + 8);
    v16h Ah = __builtin_shufflevector(ha0, ha1, 0,1,2,3,4,5,6,7,8,9,10,11,12,13,14,15);
    // B fragments from f16 weights (row-major W gives contiguous K)
    {
      const _Float16* p = WnH + (size_t)col0 * D_ + k0 + kb16;
      v8h b0 = *(const v8h*)p, b1 = *(const v8h*)(p + 8);
      v16h Bv = __builtin_shufflevector(b0, b1, 0,1,2,3,4,5,6,7,8,9,10,11,12,13,14,15);
      h0 = __builtin_amdgcn_wmma_f32_16x16x32_f16(false, An, false, Bv, (short)0, h0, false, false);
    }
    {
      const _Float16* p = WiH + (size_t)col0 * D_ + k0 + kb16;
      v8h b0 = *(const v8h*)p, b1 = *(const v8h*)(p + 8);
      v16h Bv = __builtin_shufflevector(b0, b1, 0,1,2,3,4,5,6,7,8,9,10,11,12,13,14,15);
      h0 = __builtin_amdgcn_wmma_f32_16x16x32_f16(false, Ah, false, Bv, (short)0, h0, false, false);
    }
    {
      const _Float16* p = WnH + (size_t)col1 * D_ + k0 + kb16;
      v8h b0 = *(const v8h*)p, b1 = *(const v8h*)(p + 8);
      v16h Bv = __builtin_shufflevector(b0, b1, 0,1,2,3,4,5,6,7,8,9,10,11,12,13,14,15);
      h1 = __builtin_amdgcn_wmma_f32_16x16x32_f16(false, An, false, Bv, (short)0, h1, false, false);
    }
    {
      const _Float16* p = WiH + (size_t)col1 * D_ + k0 + kb16;
      v8h b0 = *(const v8h*)p, b1 = *(const v8h*)(p + 8);
      v16h Bv = __builtin_shufflevector(b0, b1, 0,1,2,3,4,5,6,7,8,9,10,11,12,13,14,15);
      h1 = __builtin_amdgcn_wmma_f32_16x16x32_f16(false, Ah, false, Bv, (short)0, h1, false, false);
    }
  }

  // relu + dot with w_time, reduce across 16 lanes per half, then across waves
  {
    float wt0 = w_time[col0], wt1 = w_time[col1];
    float rowsum[8];
    #pragma unroll
    for (int r = 0; r < 8; ++r)
      rowsum[r] = fmaxf(h0[r], 0.0f) * wt0 + fmaxf(h1[r], 0.0f) * wt1;
    #pragma unroll
    for (int m = 1; m < 16; m <<= 1) {
      #pragma unroll
      for (int r = 0; r < 8; ++r)
        rowsum[r] += __shfl_xor(rowsum[r], m, 32);
    }
    if (lnlow == 0) {
      #pragma unroll
      for (int r = 0; r < 8; ++r)
        sOutW[wv * 16 + (hi << 3) + r] = rowsum[r];
    }
  }
  __syncthreads();
  if (tid < 16) {
    float s = 0.f;
    #pragma unroll
    for (int w = 0; w < 8; ++w) s += sOutW[w * 16 + tid];
    float x = s + b_time[0];
    float o = (x > 20.0f) ? x : log1pf(__expf(x));
    out[(size_t)b * L_ + i0 + tid] = o;
  }
}

extern "C" void kernel_launch(void* const* d_in, const int* in_sizes, int n_in,
                              void* d_out, int out_size, void* d_ws, size_t ws_size,
                              hipStream_t stream) {
  (void)in_sizes; (void)n_in; (void)out_size; (void)ws_size;
  // setup_inputs order:
  // 0 event_type(i32, unused) 1 event_time 2 noise 3 gate_params 4 length_scale
  // 5 Wn 6 Wi 7 w_time 8 b_time 9 ln_gamma 10 ln_beta
  const float* et    = (const float*)d_in[1];
  const float* noise = (const float*)d_in[2];
  const float* gp    = (const float*)d_in[3];
  const float* lsc   = (const float*)d_in[4];
  const float* Wn    = (const float*)d_in[5];
  const float* Wi    = (const float*)d_in[6];
  const float* wt    = (const float*)d_in[7];
  const float* bt    = (const float*)d_in[8];
  const float* gam   = (const float*)d_in[9];
  const float* bet   = (const float*)d_in[10];
  float* out = (float*)d_out;

  // workspace layout: embT (B*D*LP f16) | WnH (D*D f16) | WiH (D*D f16)
  _Float16* embT = (_Float16*)d_ws;
  _Float16* WnH  = embT + (size_t)B_ * D_ * LP_;
  _Float16* WiH  = WnH + (size_t)D_ * D_;

  embT_kernel<<<B_ * D_, 256, 0, stream>>>(et, embT);
  wcvt_kernel<<<(D_ * D_ + 255) / 256, 256, 0, stream>>>(Wn, Wi, WnH, WiH);
  fused_kernel<<<B_ * (L_ / 16), 256, 0, stream>>>(et, noise, gp, lsc, embT,
                                                   WnH, WiH, wt, bt, gam, bet, out);
}
